// PCGTConvLayer_9225589752432
// MI455X (gfx1250) — matrix operand order repaired and checked by reading, not compiled, not run on hardware
//
#include <hip/hip_runtime.h>
#include <hip/hip_bf16.h>
#include <stdint.h>

#define N_TOK 32768
#define IN_CH 256
#define DHEAD 64
#define NHEAD 4
#define NSEED 4
#define NPART 64
#define PSIZE 512
#define RTOT  (NPART * NSEED)
#define INV_SCALE 0.125f

typedef __bf16 bh;
typedef __attribute__((ext_vector_type(16))) __bf16 v16bh;
typedef __attribute__((ext_vector_type(8)))  float  v8f;

// ---- CDNA5 async global->LDS support (guarded; sync fallback) --------------
#if defined(__has_builtin)
#if __has_builtin(__builtin_amdgcn_global_load_async_to_lds_b128)
#define HAVE_ASYNC_LDS 1
#endif
#endif
#ifndef HAVE_ASYNC_LDS
#define HAVE_ASYNC_LDS 0
#endif

#if HAVE_ASYNC_LDS
// Param 0 of the builtin is: int __attribute__((vector_size(16))) __device__*
// (global AS int4 pointer); param 1 is the LDS-side int4 pointer.
typedef int v4i_ __attribute__((vector_size(4 * sizeof(int))));
typedef __attribute__((address_space(1))) v4i_* as1_v4i;
typedef __attribute__((address_space(3))) v4i_* as3_v4i;
#if __has_builtin(__builtin_amdgcn_s_wait_asynccnt)
#define ASYNC_WAIT0() __builtin_amdgcn_s_wait_asynccnt(0)
#else
#define ASYNC_WAIT0() asm volatile("s_wait_asynccnt 0x0" ::: "memory")
#endif
#else
#define ASYNC_WAIT0()
#endif

// copy 16 bytes global -> LDS
static __device__ __forceinline__ void cp16(const bh* g, bh* l) {
#if HAVE_ASYNC_LDS
  __builtin_amdgcn_global_load_async_to_lds_b128(
      (as1_v4i)(uintptr_t)(const void*)g,
      (as3_v4i)(unsigned)(uintptr_t)(void*)l, 0, 0);
#else
  *(uint4*)l = *(const uint4*)g;
#endif
}

static __device__ __forceinline__ int laneid() { return (int)(threadIdx.x & 31u); }

// A fragment: 16(M) x 32(K) bf16, tile stored row-major [m][k] in LDS.
static __device__ __forceinline__ v16bh load_A_rm(const bh* t, int ld) {
  int l = laneid();
  const bh* r = t + (l & 15) * ld;
  int kb = (l >> 4) << 4;
  v16bh a;
#pragma unroll
  for (int i = 0; i < 8; ++i) a[i] = r[kb + i];
#pragma unroll
  for (int i = 0; i < 8; ++i) a[8 + i] = r[kb + 8 + i];
  return a;
}

// A fragment from packed-bf16x2 tile [16 rows][16 uints]; uint c = K pair (2c,2c+1)
static __device__ __forceinline__ v16bh load_A_pk(const unsigned* t) {
  int l = laneid();
  const unsigned* r = t + (l & 15) * 16;
  int kb = (l >> 4) << 3;  // 0 or 8 (uint granularity = 2 K)
  v16bh a;
  union { unsigned u; __bf16 h[2]; } p;
#pragma unroll
  for (int i = 0; i < 4; ++i) {
    p.u = r[kb + i];
    a[2 * i] = p.h[0];
    a[2 * i + 1] = p.h[1];
  }
#pragma unroll
  for (int i = 0; i < 4; ++i) {
    p.u = r[kb + 4 + i];
    a[8 + 2 * i] = p.h[0];
    a[8 + 2 * i + 1] = p.h[1];
  }
  return a;
}

// B fragment: 32(K) x 16(N), tile stored row-major [n][k]; ld in elements
static __device__ __forceinline__ v16bh load_B_nk(const bh* t, int ld) {
  int l = laneid();
  const bh* r = t + (l & 15) * ld + ((l >> 4) << 4);
  v16bh b;
#pragma unroll
  for (int i = 0; i < 16; ++i) b[i] = r[i];
  return b;
}

// B fragment: 32(K) x 16(N), tile stored row-major [k][n]
static __device__ __forceinline__ v16bh load_B_kn(const bh* t, int ld) {
  int l = laneid();
  int n = l & 15;
  int kb = (l >> 4) << 4;
  v16bh b;
#pragma unroll
  for (int i = 0; i < 16; ++i) b[i] = t[(kb + i) * ld + n];
  return b;
}

static __device__ __forceinline__ v8f wmma_bf(v16bh a, v16bh b, v8f c) {
  return __builtin_amdgcn_wmma_f32_16x16x32_bf16(false, a, false, b, (short)0, c,
                                                 false, false);
}

static __device__ __forceinline__ unsigned pack_bf2(float a, float b) {
  union { unsigned u; __bf16 h[2]; } p;
  p.h[0] = (bh)a;
  p.h[1] = (bh)b;
  return p.u;
}

// ---------------------------------------------------------------------------
// Kernel 1: projection out = x @ W^T + b (bf16), optional x_self = mean_h V.
// ---------------------------------------------------------------------------
__global__ __launch_bounds__(256) void qkv_kernel(
    const float* __restrict__ x, const float* __restrict__ Ww,
    const float* __restrict__ Wb, bh* __restrict__ outbf,
    float* __restrict__ x_self) {
  __shared__ __align__(16) bh xs_t[128][40];
  __shared__ __align__(16) bh w_t[256][40];
  int tid = threadIdx.x;
  int wv = tid >> 5;
  int lane = tid & 31;
  int rowbase = blockIdx.x * 128;
  v8f z = {0.f, 0.f, 0.f, 0.f, 0.f, 0.f, 0.f, 0.f};
  v8f acc[16];
#pragma unroll
  for (int i = 0; i < 16; ++i) acc[i] = z;

  for (int kc = 0; kc < IN_CH / 32; ++kc) {
    {  // stage x tile 128x32 -> bf16 LDS
      int r = tid >> 1, c0 = (tid & 1) * 16;
      const float* src = x + (size_t)(rowbase + r) * IN_CH + kc * 32 + c0;
      if (kc + 1 < IN_CH / 32) __builtin_prefetch(src + 32, 0, 3);
#pragma unroll
      for (int i = 0; i < 16; ++i) xs_t[r][c0 + i] = (bh)src[i];
    }
    {  // stage W tile 256x32 -> bf16 LDS
      const float* src = Ww + (size_t)tid * IN_CH + kc * 32;
#pragma unroll
      for (int i = 0; i < 32; ++i) w_t[tid][i] = (bh)src[i];
    }
    __syncthreads();
    v16bh a = load_A_rm(&xs_t[wv * 16][0], 40);
#pragma unroll
    for (int nt = 0; nt < 16; ++nt) {
      v16bh b = load_B_nk(&w_t[nt * 16][0], 40);
      acc[nt] = wmma_bf(a, b, acc[nt]);
    }
    __syncthreads();
  }

  int hf = lane >> 4, cl = lane & 15;
#pragma unroll
  for (int nt = 0; nt < 16; ++nt) {
    float bias = Wb[nt * 16 + cl];
#pragma unroll
    for (int v = 0; v < 8; ++v) {
      float val = acc[nt][v] + bias;
      acc[nt][v] = val;
      int row = rowbase + wv * 16 + v + 8 * hf;
      outbf[(size_t)row * IN_CH + nt * 16 + cl] = (bh)val;
    }
  }
  if (x_self) {
#pragma unroll
    for (int dt = 0; dt < 4; ++dt) {
#pragma unroll
      for (int v = 0; v < 8; ++v) {
        float s = 0.25f * (acc[dt][v] + acc[dt + 4][v] + acc[dt + 8][v] +
                           acc[dt + 12][v]);
        int row = rowbase + wv * 16 + v + 8 * hf;
        x_self[(size_t)row * DHEAD + dt * 16 + cl] = s;
      }
    }
  }
}

// ---------------------------------------------------------------------------
// Kernel 2: local flash attention, double-buffered async K/V staging.
// Key permutation per 32-chunk: s0 = even keys, s1 = odd keys, so P packs
// directly into the bf16x2 A-fragment pair layout.
// ---------------------------------------------------------------------------
__global__ __launch_bounds__(256) void local_attn_kernel(
    const bh* __restrict__ Qbf, const bh* __restrict__ Kbf,
    const bh* __restrict__ Vbf, const int* __restrict__ pidx,
    float* __restrict__ out_local) {
  __shared__ __align__(16) bh Qt[128][72];
  __shared__ __align__(16) bh Kt[2][32][72];
  __shared__ __align__(16) bh Vt[2][32][72];
  __shared__ __align__(16) unsigned Pp[8][16][16];
  int p = blockIdx.x, h = blockIdx.y, rb = blockIdx.z;
  int tid = threadIdx.x, wv = tid >> 5, lane = tid & 31;
  int hf = lane >> 4, cl = lane & 15;
  const int* idx = pidx + p * PSIZE;
  int q0 = rb * 128;

  int sr = tid >> 3, sd = (tid & 7) * 8;  // K/V staging slice (16B)
  {  // stage Q tile (gathered rows), 4x16B per thread
    int r = tid >> 1, d0 = (tid & 1) * 32;
    int n = idx[q0 + r];
    const bh* src = Qbf + (size_t)n * IN_CH + h * DHEAD + d0;
#pragma unroll
    for (int i = 0; i < 4; ++i) cp16(src + i * 8, &Qt[r][d0 + i * 8]);
  }
  {  // stage chunk 0 into buffer 0
    int n = idx[sr];
    cp16(Kbf + (size_t)n * IN_CH + h * DHEAD + sd, &Kt[0][sr][sd]);
    cp16(Vbf + (size_t)n * IN_CH + h * DHEAD + sd, &Vt[0][sr][sd]);
  }
  ASYNC_WAIT0();
  __syncthreads();

  v16bh qa0 = load_A_rm(&Qt[wv * 16][0], 72);
  v16bh qa1 = load_A_rm(&Qt[wv * 16][32], 72);
  v8f z = {0.f, 0.f, 0.f, 0.f, 0.f, 0.f, 0.f, 0.f};
  v8f o[4] = {z, z, z, z};
  float m_[8], l_[8];
#pragma unroll
  for (int v = 0; v < 8; ++v) { m_[v] = -1e30f; l_[v] = 0.f; }

  for (int kc = 0; kc < PSIZE / 32; ++kc) {
    int buf = kc & 1;
    if (kc + 1 < PSIZE / 32) {  // prefetch next chunk into other buffer
      int n = idx[(kc + 1) * 32 + sr];
      cp16(Kbf + (size_t)n * IN_CH + h * DHEAD + sd, &Kt[buf ^ 1][sr][sd]);
      cp16(Vbf + (size_t)n * IN_CH + h * DHEAD + sd, &Vt[buf ^ 1][sr][sd]);
    }
    // scores: s0 = even keys (rows 0,2,..), s1 = odd keys (rows 1,3,..)
    v8f s0 = z, s1 = z;
    s0 = wmma_bf(qa0, load_B_nk(&Kt[buf][0][0], 144), s0);
    s0 = wmma_bf(qa1, load_B_nk(&Kt[buf][0][32], 144), s0);
    s1 = wmma_bf(qa0, load_B_nk(&Kt[buf][1][0], 144), s1);
    s1 = wmma_bf(qa1, load_B_nk(&Kt[buf][1][32], 144), s1);
#pragma unroll
    for (int v = 0; v < 8; ++v) {
      float a0 = s0[v] * INV_SCALE, a1 = s1[v] * INV_SCALE;
      float mx = fmaxf(a0, a1);
      mx = fmaxf(mx, __shfl_xor(mx, 1));
      mx = fmaxf(mx, __shfl_xor(mx, 2));
      mx = fmaxf(mx, __shfl_xor(mx, 4));
      mx = fmaxf(mx, __shfl_xor(mx, 8));
      float mn = fmaxf(m_[v], mx);
      float f = __expf(m_[v] - mn);
      float p0 = __expf(a0 - mn), p1 = __expf(a1 - mn);
      float rs = p0 + p1;
      rs += __shfl_xor(rs, 1);
      rs += __shfl_xor(rs, 2);
      rs += __shfl_xor(rs, 4);
      rs += __shfl_xor(rs, 8);
      l_[v] = l_[v] * f + rs;
      m_[v] = mn;
      o[0][v] *= f; o[1][v] *= f; o[2][v] *= f; o[3][v] *= f;
      // packed P: uint at col cl = keys (2cl, 2cl+1) -> A-fragment pair layout
      Pp[wv][v + 8 * hf][cl] = pack_bf2(p0, p1);
    }
    v16bh pa = load_A_pk(&Pp[wv][0][0]);  // same-wave DS in-order
#pragma unroll
    for (int j = 0; j < 4; ++j)
      o[j] = wmma_bf(pa, load_B_kn(&Vt[buf][0][j * 16], 72), o[j]);
    ASYNC_WAIT0();
    __syncthreads();
  }
#pragma unroll
  for (int v = 0; v < 8; ++v) {
    float inv = l_[v] > 0.f ? 1.f / l_[v] : 0.f;
    int n = idx[q0 + wv * 16 + v + 8 * hf];
#pragma unroll
    for (int j = 0; j < 4; ++j)
      out_local[((size_t)n * NHEAD + h) * DHEAD + j * 16 + cl] = o[j][v] * inv;
  }
}

// ---------------------------------------------------------------------------
// Kernel 3: seed pooling -> reps_k / reps_v (bf16). One block per (p,h).
// ---------------------------------------------------------------------------
__global__ __launch_bounds__(256) void pool_kernel(
    const bh* __restrict__ Kbf, const bh* __restrict__ Vbf,
    const int* __restrict__ pidx, const float* __restrict__ seeds,
    bh* __restrict__ reps_k, bh* __restrict__ reps_v) {
  __shared__ float sd[NSEED][DHEAD];
  __shared__ float lg[NSEED][PSIZE];
  __shared__ float red[NSEED][64];
  __shared__ float mxs[NSEED], sms[NSEED];
  int p = blockIdx.x, h = blockIdx.y, tid = threadIdx.x;
  const int* idx = pidx + p * PSIZE;
  {
    int m = tid >> 6, d = tid & 63;
    sd[m][d] = seeds[((size_t)m * NHEAD + h) * DHEAD + d];
  }
  __syncthreads();
  for (int s = tid; s < PSIZE; s += 256) {
    int n = idx[s];
    const bh* kr = Kbf + (size_t)n * IN_CH + h * DHEAD;
    float a0 = 0.f, a1 = 0.f, a2 = 0.f, a3 = 0.f;
#pragma unroll 8
    for (int d = 0; d < DHEAD; ++d) {
      float kv = (float)kr[d];
      a0 += sd[0][d] * kv; a1 += sd[1][d] * kv;
      a2 += sd[2][d] * kv; a3 += sd[3][d] * kv;
    }
    lg[0][s] = a0 * INV_SCALE; lg[1][s] = a1 * INV_SCALE;
    lg[2][s] = a2 * INV_SCALE; lg[3][s] = a3 * INV_SCALE;
  }
  __syncthreads();
  int m = tid >> 6, i = tid & 63;
  {
    float mx = -1e30f;
    for (int s = i; s < PSIZE; s += 64) mx = fmaxf(mx, lg[m][s]);
    red[m][i] = mx;
  }
  __syncthreads();
  if (tid < NSEED) {
    float mx = -1e30f;
    for (int k = 0; k < 64; ++k) mx = fmaxf(mx, red[tid][k]);
    mxs[tid] = mx;
  }
  __syncthreads();
  {
    float sm = 0.f, mx = mxs[m];
    for (int s = i; s < PSIZE; s += 64) {
      float e = __expf(lg[m][s] - mx);
      lg[m][s] = e;
      sm += e;
    }
    red[m][i] = sm;
  }
  __syncthreads();
  if (tid < NSEED) {
    float sm = 0.f;
    for (int k = 0; k < 64; ++k) sm += red[tid][k];
    sms[tid] = sm;
  }
  __syncthreads();
  {
    int d = i;
    float ak = 0.f, av = 0.f;
    for (int s = 0; s < PSIZE; ++s) {
      int n = idx[s];
      float wgt = lg[m][s];
      ak += wgt * (float)Kbf[(size_t)n * IN_CH + h * DHEAD + d];
      av += wgt * (float)Vbf[(size_t)n * IN_CH + h * DHEAD + d];
    }
    float inv = 1.f / sms[m];
    size_t r = ((size_t)(p * NSEED + m) * NHEAD + h) * DHEAD + d;
    reps_k[r] = (bh)(ak * inv);
    reps_v[r] = (bh)(av * inv);
  }
}

// ---------------------------------------------------------------------------
// Kernel 4: global cross-attention over 256 reps + final gated combine.
// ---------------------------------------------------------------------------
__global__ __launch_bounds__(256) void cross_kernel(
    const bh* __restrict__ Qbf, const bh* __restrict__ reps_k,
    const bh* __restrict__ reps_v, const float* __restrict__ out_local,
    const float* __restrict__ x_self, const float* __restrict__ bscore,
    const float* __restrict__ alpha_logit, const float* __restrict__ bweight,
    const float* __restrict__ beta_logit, float* __restrict__ out) {
  __shared__ __align__(16) bh Qt[128][72];
  __shared__ __align__(16) bh Kt[32][72];
  __shared__ __align__(16) bh Vt[32][72];
  __shared__ __align__(16) unsigned Pp[8][16][16];
  int tid = threadIdx.x, wv = tid >> 5, lane = tid & 31;
  int hf = lane >> 4, cl = lane & 15;
  int nbase = blockIdx.x * 128;
  int sr = tid >> 3, sd = (tid & 7) * 8;
  v8f z = {0.f, 0.f, 0.f, 0.f, 0.f, 0.f, 0.f, 0.f};
  v8f xg[4] = {z, z, z, z};

  for (int h = 0; h < NHEAD; ++h) {
    __syncthreads();
    {  // stage Q tile for this head
      int r = tid >> 1, d0 = (tid & 1) * 32;
      const bh* src = Qbf + (size_t)(nbase + r) * IN_CH + h * DHEAD + d0;
#pragma unroll
      for (int i = 0; i < 4; ++i) cp16(src + i * 8, &Qt[r][d0 + i * 8]);
    }
    ASYNC_WAIT0();
    __syncthreads();
    v16bh qa0 = load_A_rm(&Qt[wv * 16][0], 72);
    v16bh qa1 = load_A_rm(&Qt[wv * 16][32], 72);
    v8f o[4] = {z, z, z, z};
    float m_[8], l_[8];
#pragma unroll
    for (int v = 0; v < 8; ++v) { m_[v] = -1e30f; l_[v] = 0.f; }

    for (int kc = 0; kc < RTOT / 32; ++kc) {
      __syncthreads();
      {
        int rep = kc * 32 + sr;
        cp16(reps_k + ((size_t)rep * NHEAD + h) * DHEAD + sd, &Kt[sr][sd]);
        cp16(reps_v + ((size_t)rep * NHEAD + h) * DHEAD + sd, &Vt[sr][sd]);
      }
      ASYNC_WAIT0();
      __syncthreads();
      v8f s0 = z, s1 = z;
      s0 = wmma_bf(qa0, load_B_nk(&Kt[0][0], 144), s0);
      s0 = wmma_bf(qa1, load_B_nk(&Kt[0][32], 144), s0);
      s1 = wmma_bf(qa0, load_B_nk(&Kt[1][0], 144), s1);
      s1 = wmma_bf(qa1, load_B_nk(&Kt[1][32], 144), s1);
#pragma unroll
      for (int v = 0; v < 8; ++v) {
        float a0 = s0[v] * INV_SCALE, a1 = s1[v] * INV_SCALE;
        float mx = fmaxf(a0, a1);
        mx = fmaxf(mx, __shfl_xor(mx, 1));
        mx = fmaxf(mx, __shfl_xor(mx, 2));
        mx = fmaxf(mx, __shfl_xor(mx, 4));
        mx = fmaxf(mx, __shfl_xor(mx, 8));
        float mn = fmaxf(m_[v], mx);
        float f = __expf(m_[v] - mn);
        float p0 = __expf(a0 - mn), p1 = __expf(a1 - mn);
        float rs = p0 + p1;
        rs += __shfl_xor(rs, 1);
        rs += __shfl_xor(rs, 2);
        rs += __shfl_xor(rs, 4);
        rs += __shfl_xor(rs, 8);
        l_[v] = l_[v] * f + rs;
        m_[v] = mn;
        o[0][v] *= f; o[1][v] *= f; o[2][v] *= f; o[3][v] *= f;
        Pp[wv][v + 8 * hf][cl] = pack_bf2(p0, p1);
      }
      v16bh pa = load_A_pk(&Pp[wv][0][0]);
#pragma unroll
      for (int j = 0; j < 4; ++j)
        o[j] = wmma_bf(pa, load_B_kn(&Vt[0][j * 16], 72), o[j]);
    }
#pragma unroll
    for (int v = 0; v < 8; ++v) {
      float inv = l_[v] > 0.f ? 1.f / l_[v] : 0.f;
#pragma unroll
      for (int j = 0; j < 4; ++j) xg[j][v] += o[j][v] * inv;
    }
  }

  float beta = 2.f / (1.f + __expf(-beta_logit[0]));
  float al = alpha_logit[0], bw = bweight[0];
#pragma unroll
  for (int v = 0; v < 8; ++v) {
    int n = nbase + wv * 16 + v + 8 * hf;
    float alpha = 1.f / (1.f + __expf(-(al + bw * bscore[n])));
#pragma unroll
    for (int j = 0; j < 4; ++j) {
      int d = j * 16 + cl;
      const float* olr = out_local + (size_t)n * NHEAD * DHEAD + d;
      float xl = 0.25f * (olr[0] + olr[DHEAD] + olr[2 * DHEAD] + olr[3 * DHEAD]);
      float xs = x_self[(size_t)n * DHEAD + d];
      float g = 0.25f * xg[j][v];
      out[(size_t)n * DHEAD + d] = alpha * xl + (1.f - alpha) * g + beta * xs;
    }
  }
}

// ---------------------------------------------------------------------------
extern "C" void kernel_launch(void* const* d_in, const int* in_sizes, int n_in,
                              void* d_out, int out_size, void* d_ws,
                              size_t ws_size, hipStream_t stream) {
  const float* x = (const float*)d_in[0];
  const int* pidx = (const int*)d_in[1];
  const float* bscore = (const float*)d_in[2];
  const float* Wq_w = (const float*)d_in[3];
  const float* Wq_b = (const float*)d_in[4];
  const float* Wk_w = (const float*)d_in[5];
  const float* Wk_b = (const float*)d_in[6];
  const float* Wv_w = (const float*)d_in[7];
  const float* Wv_b = (const float*)d_in[8];
  const float* seeds = (const float*)d_in[9];
  const float* alpha_logit = (const float*)d_in[10];
  const float* bweight = (const float*)d_in[11];
  const float* beta_logit = (const float*)d_in[12];

  char* ws = (char*)d_ws;
  size_t off = 0;
  bh* Qbf = (bh*)(ws + off); off += (size_t)N_TOK * IN_CH * sizeof(bh);
  bh* Kbf = (bh*)(ws + off); off += (size_t)N_TOK * IN_CH * sizeof(bh);
  bh* Vbf = (bh*)(ws + off); off += (size_t)N_TOK * IN_CH * sizeof(bh);
  float* x_self = (float*)(ws + off); off += (size_t)N_TOK * DHEAD * sizeof(float);
  float* out_local = (float*)(ws + off);
  off += (size_t)N_TOK * NHEAD * DHEAD * sizeof(float);
  bh* reps_k = (bh*)(ws + off); off += (size_t)RTOT * NHEAD * DHEAD * sizeof(bh);
  bh* reps_v = (bh*)(ws + off); off += (size_t)RTOT * NHEAD * DHEAD * sizeof(bh);

  dim3 blk(256);
  qkv_kernel<<<dim3(N_TOK / 128), blk, 0, stream>>>(x, Wq_w, Wq_b, Qbf, nullptr);
  qkv_kernel<<<dim3(N_TOK / 128), blk, 0, stream>>>(x, Wk_w, Wk_b, Kbf, nullptr);
  qkv_kernel<<<dim3(N_TOK / 128), blk, 0, stream>>>(x, Wv_w, Wv_b, Vbf, x_self);
  local_attn_kernel<<<dim3(NPART, NHEAD, PSIZE / 128), blk, 0, stream>>>(
      Qbf, Kbf, Vbf, pidx, out_local);
  pool_kernel<<<dim3(NPART, NHEAD), blk, 0, stream>>>(Kbf, Vbf, pidx, seeds,
                                                      reps_k, reps_v);
  cross_kernel<<<dim3(N_TOK / 128), blk, 0, stream>>>(
      Qbf, reps_k, reps_v, out_local, x_self, bscore, alpha_logit, bweight,
      beta_logit, (float*)d_out);
}
